// MultiHeadRecurrentAttention_73263552135473
// MI455X (gfx1250) — compile-verified, matrix operand order checked by reading, not run on hardware
//
#include <hip/hip_runtime.h>
#include <hip/hip_bf16.h>

typedef __attribute__((ext_vector_type(16))) __bf16 v16bf;
typedef __attribute__((ext_vector_type(8)))  __bf16 v8bf;
typedef __attribute__((ext_vector_type(8)))  float  v8f;

#define BB 2
#define LL 128
#define FF 1024
#define HH 8
#define DD 128
#define HD 1024   // H*D
#define G4 512    // 4*D

// ---------------- WMMA helpers ----------------------------------------------
// 16-bit A-matrix 16x32 layout (ISA 7.12.2): lane = half*16 + r holds row r,
// VGPR j<4 -> K = 2j + 8*half (pairs), VGPR j>=4 -> K = 16 + 2(j-4) + 8*half.
// => per-lane data is A[row, half*8 .. +8] and A[row, 16+half*8 .. +8]:
// two contiguous 16-byte loads. B operand uses the same K packing with lanes
// indexing N, so a transposed (NxK) weight row loads identically.
__device__ __forceinline__ v16bf ld_frag(const __bf16* p) {
  v8bf lo = *(const v8bf*)p;
  v8bf hi = *(const v8bf*)(p + 16);
  v16bf r;
#pragma unroll
  for (int i = 0; i < 8; ++i) { r[i] = lo[i]; r[i + 8] = hi[i]; }
  return r;
}

__device__ __forceinline__ v8f wmma_bf16(v16bf a, v16bf b, v8f c) {
  return __builtin_amdgcn_wmma_f32_16x16x32_bf16(false, a, false, b, (short)0, c,
                                                 false, false);
}

// C = A(MxK, row-major bf16) * Bt(NxK, row-major bf16)^T for one 16x16 tile.
__device__ __forceinline__ v8f gemm_tile(const __bf16* __restrict__ A,
                                         const __bf16* __restrict__ Bt,
                                         int lda, int ldb, int K,
                                         int m0, int n0, int lane) {
  int r = lane & 15, half = lane >> 4;
  const __bf16* arow = A + (size_t)(m0 + r) * lda + half * 8;
  const __bf16* brow = Bt + (size_t)(n0 + r) * ldb + half * 8;
  v8f acc = {};
  for (int kb = 0; kb < K; kb += 32) {
    __builtin_prefetch(arow + kb + 128, 0, 1);
    v16bf af = ld_frag(arow + kb);
    v16bf bf = ld_frag(brow + kb);
    acc = wmma_bf16(af, bf, acc);
  }
  return acc;
}

__device__ __forceinline__ float sigm(float x) { return 1.0f / (1.0f + __expf(-x)); }
__device__ __forceinline__ float tanh_ap(float x) { return 2.0f * sigm(2.0f * x) - 1.0f; }

// ---------------- setup kernels ---------------------------------------------
__global__ void k_cvt_bf16(const float* __restrict__ in, __bf16* __restrict__ out, int n) {
  int i = blockIdx.x * blockDim.x + threadIdx.x;
  if (i < n) out[i] = (__bf16)in[i];
}

// (batch, K, N) fp32 -> (batch, N, K) bf16
__global__ void k_transpose_bf16(const float* __restrict__ in, __bf16* __restrict__ out,
                                 int batch, int K, int N) {
  int i = blockIdx.x * blockDim.x + threadIdx.x;
  int total = batch * K * N;
  if (i >= total) return;
  int n = i % N;
  int t = i / N;
  int kk = t % K;
  int b = t / K;
  out[(size_t)b * N * K + (size_t)n * K + kk] = (__bf16)in[i];
}

// ---------------- QKV projection: (256x1024)@(1024x1024)+bias --------------
// Writes bf16 head-split (B,H,L,D).
__global__ void k_gemm_qkv(const __bf16* __restrict__ Abf, const __bf16* __restrict__ Wt,
                           const float* __restrict__ bias, __bf16* __restrict__ outHS) {
  int wv = threadIdx.x >> 5, lane = threadIdx.x & 31;
  int waveId = blockIdx.x * 8 + wv;   // 1024 wave tiles
  int tm = waveId >> 6;               // M=256 -> 16 tiles
  int tn = waveId & 63;               // N=1024 -> 64 tiles
  int m0 = tm * 16, n0 = tn * 16;
  v8f acc = gemm_tile(Abf, Wt, FF, FF, FF, m0, n0, lane);
  int n = lane & 15, half = lane >> 4;
  int col = n0 + n;
  float bv = bias[col];
  int h = col >> 7, d = col & 127;
#pragma unroll
  for (int j = 0; j < 8; ++j) {
    int gm = m0 + j + 8 * half;
    int b = gm >> 7, l = gm & 127;
    outHS[(size_t)((b * HH + h) * LL + l) * DD + d] = (__bf16)(acc[j] + bv);
  }
}

// ---------------- VX = v @ Wx per (b,h): (128x128)@(128x512) ----------------
__global__ void k_gemm_vx(const __bf16* __restrict__ vHS, const __bf16* __restrict__ WxT,
                          float* __restrict__ VX) {
  int wv = threadIdx.x >> 5, lane = threadIdx.x & 31;
  int waveId = blockIdx.x * 8 + wv;   // 4096 wave tiles
  int bh = waveId >> 8;
  int t = waveId & 255;
  int tm = t >> 5, tn = t & 31;       // 8 x 32 tiles
  int m0 = tm * 16, n0 = tn * 16;
  const __bf16* A = vHS + (size_t)bh * LL * DD;
  const __bf16* Bt = WxT + (size_t)(bh & 7) * G4 * DD;
  v8f acc = gemm_tile(A, Bt, DD, DD, DD, m0, n0, lane);
  int n = lane & 15, half = lane >> 4;
  int col = n0 + n;
#pragma unroll
  for (int j = 0; j < 8; ++j) {
    int l = m0 + j + 8 * half;
    VX[((size_t)bh * LL + l) * G4 + col] = acc[j];
  }
}

// ---------------- logits = q @ k^T / sqrt(D) per (b,h) ----------------------
__global__ void k_logits(const __bf16* __restrict__ qHS, const __bf16* __restrict__ kHS,
                         float* __restrict__ logits) {
  int wv = threadIdx.x >> 5, lane = threadIdx.x & 31;
  int waveId = blockIdx.x * 8 + wv;   // 1024 wave tiles
  int bh = waveId >> 6;
  int t = waveId & 63;
  int tm = t >> 3, tn = t & 7;
  int m0 = tm * 16, n0 = tn * 16;
  const __bf16* A = qHS + (size_t)bh * LL * DD;
  const __bf16* Bt = kHS + (size_t)bh * LL * DD;  // k rows are already K-major
  v8f acc = gemm_tile(A, Bt, DD, DD, DD, m0, n0, lane);
  const float scale = 0.08838834764831845f;       // 1/sqrt(128)
  int n = lane & 15, half = lane >> 4;
#pragma unroll
  for (int j = 0; j < 8; ++j) {
    int q = m0 + j + 8 * half;
    logits[((size_t)bh * LL + q) * LL + n0 + n] = acc[j] * scale;
  }
}

// ---------------- masked softmax, one wave per row --------------------------
__global__ void k_softmax(const float* __restrict__ logits, const int* __restrict__ mask,
                          float* __restrict__ attn) {
  int wv = threadIdx.x >> 5, lane = threadIdx.x & 31;
  int row = blockIdx.x * 8 + wv;      // 2048 rows = bh*128+q
  int bh = row >> 7, q = row & 127;
  int b = bh >> 3;
  const float4* lr = (const float4*)(logits + (size_t)row * LL);
  const int4* mr = (const int4*)(mask + ((size_t)b * LL + q) * LL);
  float4 x = lr[lane];
  int4 m = mr[lane];
  float ninf = -__builtin_inff();
  float v0 = m.x ? x.x : ninf;
  float v1 = m.y ? x.y : ninf;
  float v2 = m.z ? x.z : ninf;
  float v3 = m.w ? x.w : ninf;
  float mx = fmaxf(fmaxf(v0, v1), fmaxf(v2, v3));
#pragma unroll
  for (int o = 16; o > 0; o >>= 1) mx = fmaxf(mx, __shfl_xor(mx, o, 32));
  float e0 = __expf(v0 - mx), e1 = __expf(v1 - mx);
  float e2 = __expf(v2 - mx), e3 = __expf(v3 - mx);
  float s = e0 + e1 + e2 + e3;
#pragma unroll
  for (int o = 16; o > 0; o >>= 1) s += __shfl_xor(s, o, 32);
  float inv = 1.0f / s;
  float4 ov;
  ov.x = e0 * inv; ov.y = e1 * inv; ov.z = e2 * inv; ov.w = e3 * inv;
  ((float4*)(attn + (size_t)row * LL))[lane] = ov;
}

// ---------------- persistent LSTM scan, one workgroup per (b,h) -------------
// LDS: Wh^T (512x128 bf16, 128KB) + double-buffered bf16 h (64KB) +
//      fp32 c (64KB) + bias/vx-row/attn-col (~4.5KB) ~= 261KB of the 320KB WGP LDS.
#define SM_WHL   0
#define SM_HBUF  (SM_WHL + G4 * DD * 2)           // 131072
#define SM_CST   (SM_HBUF + 2 * LL * DD * 2)      // +65536
#define SM_BIAS  (SM_CST + LL * DD * 4)           // +65536
#define SM_VXR   (SM_BIAS + G4 * 4)
#define SM_ATT   (SM_VXR + G4 * 4)
#define SM_TOTAL (SM_ATT + LL * 4)

__global__ void __launch_bounds__(512) k_lstm(const float* __restrict__ attn,
                                              const float* __restrict__ VX,
                                              const __bf16* __restrict__ WhT,
                                              const float* __restrict__ b_lstm,
                                              __bf16* __restrict__ rdpa) {
  extern __shared__ char smem[];
  __bf16* whl = (__bf16*)(smem + SM_WHL);
  __bf16* hbuf = (__bf16*)(smem + SM_HBUF);
  float* cst = (float*)(smem + SM_CST);
  float* biasl = (float*)(smem + SM_BIAS);
  float* vxr = (float*)(smem + SM_VXR);
  float* attc = (float*)(smem + SM_ATT);

  int tid = threadIdx.x;
  int bh = blockIdx.x;
  int b = bh >> 3, h = bh & 7;

  { // stage Wh^T[h] into LDS (reused 128 times)
    const v8bf* src = (const v8bf*)(WhT + (size_t)h * G4 * DD);
    v8bf* dst = (v8bf*)whl;
    for (int i = tid; i < (G4 * DD) / 8; i += 512) dst[i] = src[i];
  }
  { // zero h (both buffers) and c
    int4 z = {0, 0, 0, 0};
    int4* hz = (int4*)hbuf;
    for (int i = tid; i < (2 * LL * DD * 2) / 16; i += 512) hz[i] = z;
    int4* cz = (int4*)cst;
    for (int i = tid; i < (LL * DD * 4) / 16; i += 512) cz[i] = z;
  }
  biasl[tid] = b_lstm[h * G4 + tid];
  __syncthreads();

  int wv = tid >> 5, lane = tid & 31;
  int r = lane & 15, half = lane >> 4, n = lane & 15;
  int cur = 0;

  for (int k = 0; k < LL; ++k) {
    if (tid < 128) attc[tid] = attn[((size_t)bh * LL + tid) * LL + k];
    vxr[tid] = VX[((size_t)bh * LL + k) * G4 + tid];
    __syncthreads();

    const __bf16* hb = hbuf + cur * LL * DD;
    __bf16* hout = hbuf + (cur ^ 1) * LL * DD;

#pragma unroll
    for (int pi = 0; pi < 4; ++pi) {
      int p = wv * 4 + pi;          // static (q,d)-tile ownership across steps
      int qt = p >> 3, dt = p & 7;
      int m0 = qt * 16;
      const __bf16* arow = hb + (m0 + r) * DD + half * 8;
      v8f ai = {}, af_ = {}, ag = {}, ao = {};
#pragma unroll
      for (int kb = 0; kb < DD; kb += 32) {
        v16bf hf = ld_frag(arow + kb);
        v16bf b0 = ld_frag(whl + (0 * DD + dt * 16 + r) * DD + half * 8 + kb);
        v16bf b1 = ld_frag(whl + (1 * DD + dt * 16 + r) * DD + half * 8 + kb);
        v16bf b2 = ld_frag(whl + (2 * DD + dt * 16 + r) * DD + half * 8 + kb);
        v16bf b3 = ld_frag(whl + (3 * DD + dt * 16 + r) * DD + half * 8 + kb);
        ai = wmma_bf16(hf, b0, ai);
        af_ = wmma_bf16(hf, b1, af_);
        ag = wmma_bf16(hf, b2, ag);
        ao = wmma_bf16(hf, b3, ao);
      }
      int u0 = dt * 16 + n;         // d within head dim
#pragma unroll
      for (int j = 0; j < 8; ++j) {
        int q = m0 + j + 8 * half;
        float aw = attc[q];
        float zi = ai[j] + biasl[u0] + aw * vxr[u0];
        float zf = af_[j] + biasl[128 + u0] + aw * vxr[128 + u0];
        float zg = ag[j] + biasl[256 + u0] + aw * vxr[256 + u0];
        float zo = ao[j] + biasl[384 + u0] + aw * vxr[384 + u0];
        float co = cst[q * DD + u0];
        float cn = sigm(zf) * co + sigm(zi) * tanh_ap(zg);
        float hv = sigm(zo) * tanh_ap(cn);
        cst[q * DD + u0] = cn;
        hout[q * DD + u0] = (__bf16)hv;
        if (k == LL - 1)
          rdpa[((size_t)b * LL + q) * HD + h * DD + u0] = (__bf16)hv;
      }
    }
    __syncthreads();
    cur ^= 1;
  }
}

// ---------------- output projection: (256x1024)@(1024x128)+bo ---------------
__global__ void k_gemm_out(const __bf16* __restrict__ rdpa, const __bf16* __restrict__ WoT,
                           const float* __restrict__ bo, float* __restrict__ out) {
  int wv = threadIdx.x >> 5, lane = threadIdx.x & 31;
  int waveId = blockIdx.x * 8 + wv;   // 128 wave tiles
  int tm = waveId >> 3, tn = waveId & 7;
  int m0 = tm * 16, n0 = tn * 16;
  v8f acc = gemm_tile(rdpa, WoT, HD, HD, HD, m0, n0, lane);
  int n = lane & 15, half = lane >> 4;
  int col = n0 + n;
  float bv = bo[col];
#pragma unroll
  for (int j = 0; j < 8; ++j) {
    int gm = m0 + j + 8 * half;
    out[(size_t)gm * DD + col] = acc[j] + bv;
  }
}

// ---------------- host driver ----------------------------------------------
extern "C" void kernel_launch(void* const* d_in, const int* in_sizes, int n_in,
                              void* d_out, int out_size, void* d_ws, size_t ws_size,
                              hipStream_t stream) {
  const float* qg = (const float*)d_in[0];
  const float* kg = (const float*)d_in[1];
  const float* vg = (const float*)d_in[2];
  const int* mask = (const int*)d_in[3];
  const float* Wq = (const float*)d_in[4];
  const float* bq = (const float*)d_in[5];
  const float* Wk = (const float*)d_in[6];
  const float* bk = (const float*)d_in[7];
  const float* Wv = (const float*)d_in[8];
  const float* bv = (const float*)d_in[9];
  const float* Wx = (const float*)d_in[10];
  const float* Wh = (const float*)d_in[11];
  const float* bl = (const float*)d_in[12];
  const float* Wo = (const float*)d_in[13];
  const float* bo = (const float*)d_in[14];

  float* out = (float*)d_out;                 // (B,L,D) = 32768 floats
  float* attn = out + BB * LL * DD;           // (B,H,L,L) = 262144 floats

  char* ws = (char*)d_ws;
  size_t off = 0;
  auto alloc = [&](size_t bytes) -> void* {
    void* p = ws + off;
    off = (off + bytes + 255) & ~(size_t)255;
    return p;
  };

  __bf16* qg_bf = (__bf16*)alloc(BB * LL * FF * 2);
  __bf16* kg_bf = (__bf16*)alloc(BB * LL * FF * 2);
  __bf16* vg_bf = (__bf16*)alloc(BB * LL * FF * 2);
  __bf16* WqT = (__bf16*)alloc(FF * HD * 2);
  __bf16* WkT = (__bf16*)alloc(FF * HD * 2);
  __bf16* WvT = (__bf16*)alloc(FF * HD * 2);
  __bf16* WxT = (__bf16*)alloc(HH * G4 * DD * 2);
  __bf16* WhT = (__bf16*)alloc(HH * G4 * DD * 2);
  __bf16* WoT = (__bf16*)alloc(DD * HD * 2);
  __bf16* qHS = (__bf16*)alloc(BB * HH * LL * DD * 2);
  __bf16* kHS = (__bf16*)alloc(BB * HH * LL * DD * 2);
  __bf16* vHS = (__bf16*)alloc(BB * HH * LL * DD * 2);
  float* VX = (float*)alloc((size_t)BB * HH * LL * G4 * 4);
  float* logits = (float*)alloc((size_t)BB * HH * LL * LL * 4);
  __bf16* rdpa = (__bf16*)alloc(BB * LL * HD * 2);

  // 1) inputs -> bf16
  int nin = BB * LL * FF;
  k_cvt_bf16<<<(nin + 255) / 256, 256, 0, stream>>>(qg, qg_bf, nin);
  k_cvt_bf16<<<(nin + 255) / 256, 256, 0, stream>>>(kg, kg_bf, nin);
  k_cvt_bf16<<<(nin + 255) / 256, 256, 0, stream>>>(vg, vg_bf, nin);

  // 2) weights -> transposed bf16
  k_transpose_bf16<<<(FF * HD + 255) / 256, 256, 0, stream>>>(Wq, WqT, 1, FF, HD);
  k_transpose_bf16<<<(FF * HD + 255) / 256, 256, 0, stream>>>(Wk, WkT, 1, FF, HD);
  k_transpose_bf16<<<(FF * HD + 255) / 256, 256, 0, stream>>>(Wv, WvT, 1, FF, HD);
  k_transpose_bf16<<<(HH * DD * G4 + 255) / 256, 256, 0, stream>>>(Wx, WxT, HH, DD, G4);
  k_transpose_bf16<<<(HH * DD * G4 + 255) / 256, 256, 0, stream>>>(Wh, WhT, HH, DD, G4);
  k_transpose_bf16<<<(HD * DD + 255) / 256, 256, 0, stream>>>(Wo, WoT, 1, HD, DD);

  // 3) Q/K/V projections (WMMA)
  k_gemm_qkv<<<128, 256, 0, stream>>>(qg_bf, WqT, bq, qHS);
  k_gemm_qkv<<<128, 256, 0, stream>>>(kg_bf, WkT, bk, kHS);
  k_gemm_qkv<<<128, 256, 0, stream>>>(vg_bf, WvT, bv, vHS);

  // 4) VX = v @ Wx (rank-1 collapse of the LSTM x-path)
  k_gemm_vx<<<512, 256, 0, stream>>>(vHS, WxT, VX);

  // 5) attention logits + softmax (attn lands directly in d_out)
  k_logits<<<128, 256, 0, stream>>>(qHS, kHS, logits);
  k_softmax<<<256, 256, 0, stream>>>(logits, mask, attn);

  // 6) sequential LSTM over key axis, LDS-resident state, WMMA recurrence
  k_lstm<<<BB * HH, 512, SM_TOTAL, stream>>>(attn, VX, WhT, bl, rdpa);

  // 7) output projection
  k_gemm_out<<<16, 256, 0, stream>>>(rdpa, WoT, bo, out);
}
